// LayerWithAttention_61022895342038
// MI455X (gfx1250) — matrix operand order repaired and checked by reading, not compiled, not run on hardware
//
#include <hip/hip_runtime.h>
#include <stdint.h>

// ---------------------------------------------------------------------------
// Types for CDNA5 WMMA (gfx1250, wave32)
// ---------------------------------------------------------------------------
typedef __bf16 bf16;
typedef __attribute__((ext_vector_type(16))) __bf16 v16bf;
typedef __attribute__((ext_vector_type(8)))  float  v8f;

union FragU { v16bf v; uint4 q[2]; };

__device__ inline v8f vzero8() {
    v8f z = {0.f,0.f,0.f,0.f,0.f,0.f,0.f,0.f};
    return z;
}

// A-matrix fragment (16x32 bf16). LDS layout: [row][32] contiguous.
// lane<16: K 0..7 then 16..23 ; lane>=16: K 8..15 then 24..31 (ISA 7.12.2).
__device__ inline v16bf load_frag_a(const bf16* lds, int rowBase) {
    int lane = threadIdx.x & 31;
    int r    = rowBase + (lane & 15);
    int koff = (lane < 16) ? 0 : 8;
    FragU f;
    f.q[0] = *(const uint4*)(lds + r * 32 + koff);
    f.q[1] = *(const uint4*)(lds + r * 32 + koff + 16);
    return f.v;
}

// B-matrix fragment (32x16 bf16), staged column-major: LDS [col][32].
// lane<16: K 0..7 then 16..23 ; lane>=16: K 8..15 then 24..31.
__device__ inline v16bf load_frag_b(const bf16* lds, int colBase) {
    int lane = threadIdx.x & 31;
    int c    = colBase + (lane & 15);
    int koff = (lane < 16) ? 0 : 8;
    FragU f;
    f.q[0] = *(const uint4*)(lds + c * 32 + koff);
    f.q[1] = *(const uint4*)(lds + c * 32 + koff + 16);
    return f.v;
}

#define WMMA_BF16(acc, a, b)                                                   \
    acc = __builtin_amdgcn_wmma_f32_16x16x32_bf16(false, a, false, b,          \
                                                  (short)0, acc, false, false)

// ---------------------------------------------------------------------------
// CDNA5 async global->LDS copy (Tensor/async path, ASYNCcnt-tracked).
// vdst VGPR carries the LDS byte offset; vaddr the 64-bit global address.
// ---------------------------------------------------------------------------
__device__ inline void async_ld_b128(void* ldsPtr, const void* gPtr) {
    unsigned lds = (unsigned)(size_t)ldsPtr;              // low 32 bits = LDS offset
    unsigned long long g = (unsigned long long)(size_t)gPtr;
    asm volatile("global_load_async_to_lds_b128 %0, %1, off"
                 :: "v"(lds), "v"(g)
                 : "memory");
}
__device__ inline void async_wait0() {
    asm volatile("s_wait_asynccnt 0x0" ::: "memory");
}

// ---------------------------------------------------------------------------
// Problem constants
// ---------------------------------------------------------------------------
#define BATCH 16
#define TSTEP 4
#define BT    64          // BATCH*TSTEP
#define CIN   64
#define COUT  128
#define NSP   1024        // H*W
#define NROW  65536       // BT*NSP
#define EPSV  1e-5f

// ---------------------------------------------------------------------------
// K1: repack conv weights -> Wcv[kc=18][co=128][kk=32] bf16,
//     K ordering k = (kh*3+kw)*64 + ci  (chunk = fixed tap, 32 channels)
// ---------------------------------------------------------------------------
__global__ void convw_kernel(const float* __restrict__ cw, bf16* __restrict__ Wcv) {
    int idx = blockIdx.x * 256 + threadIdx.x;
    if (idx >= 18 * 128 * 32) return;
    int kc  = idx >> 12;          // /4096
    int rem = idx & 4095;
    int co  = rem >> 5;
    int kk  = rem & 31;
    int khw = kc >> 1;            // kh*3+kw
    int ci  = ((kc & 1) << 5) + kk;
    Wcv[idx] = (bf16)cw[(co * CIN + ci) * 9 + khw];
}

// ---------------------------------------------------------------------------
// K2: implicit-GEMM 3x3 conv (pad 1).  out[bt][pos][co] fp32 (no bias: BN absorbs it)
//     block = 256 thr (8 waves); tile = 64 spatial x 128 channels; K = 576
//     Weights staged via async global->LDS; im2col converted fp32->bf16 in VALU.
// ---------------------------------------------------------------------------
__global__ __launch_bounds__(256) void conv3x3_kernel(const float* __restrict__ x,
                                                      const bf16* __restrict__ Wcv,
                                                      float* __restrict__ outp) {
    int bt = blockIdx.x;          // image
    int mt = blockIdx.y;          // 0..15 : rows [mt*64, mt*64+64)
    __shared__ __align__(16) bf16 ldsA[64 * 32];
    __shared__ __align__(16) bf16 ldsB[128 * 32];
    int tid  = threadIdx.x;
    int wave = tid >> 5, lane = tid & 31;
    int msub  = wave & 3;
    int nbase = (wave >> 2) * 64;
    v8f acc[4];
    for (int j = 0; j < 4; ++j) acc[j] = vzero8();
    int h0 = mt * 2;
    const float* xin = x + (size_t)bt * (CIN * NSP);
#pragma unroll
    for (int kc = 0; kc < 18; ++kc) {
        const int khw = kc >> 1;          // compile-time after unroll
        const int kh = khw / 3, kw = khw % 3;
        const int ci0 = (kc & 1) << 5;
        // stage weights: 8KB contiguous via async global->LDS (2 x b128 per lane)
        const char* wsrc = (const char*)(Wcv + kc * 4096);
        async_ld_b128((char*)ldsB + tid * 16,         wsrc + tid * 16);
        async_ld_b128((char*)ldsB + (tid + 256) * 16, wsrc + (tid + 256) * 16);
        // stage im2col A tile (64 x 32): fp32 -> bf16 conversion (VALU path)
#pragma unroll
        for (int u = 0; u < 8; ++u) {
            int i  = tid + u * 256;
            int kk = i >> 6;
            int m  = i & 63;
            int h  = h0 + (m >> 5) + kh - 1;
            int w  = (m & 31) + kw - 1;
            float v = 0.f;
            if ((unsigned)h < 32u && (unsigned)w < 32u)
                v = xin[(ci0 + kk) * NSP + h * 32 + w];
            ldsA[m * 32 + kk] = (bf16)v;
        }
        async_wait0();
        __syncthreads();
        v16bf a = load_frag_a(ldsA, msub * 16);
#pragma unroll
        for (int j = 0; j < 4; ++j) {
            v16bf b = load_frag_b(ldsB, nbase + j * 16);
            WMMA_BF16(acc[j], a, b);
        }
        __syncthreads();
    }
    int mrow = msub * 16 + 8 * (lane >> 4);
    int ncol = lane & 15;
    float* op = outp + ((size_t)bt * NSP + mt * 64) * COUT;
#pragma unroll
    for (int j = 0; j < 4; ++j)
        for (int r = 0; r < 8; ++r)
            op[(mrow + r) * COUT + nbase + j * 16 + ncol] = acc[j][r];
}

// ---------------------------------------------------------------------------
// K3: per-channel sum / sum-of-squares over 65536 rows (atomic accumulate)
// ---------------------------------------------------------------------------
__global__ __launch_bounds__(256) void stats_kernel(const float* __restrict__ conv,
                                                    float* __restrict__ gsum,
                                                    float* __restrict__ gsq) {
    int tid  = threadIdx.x;
    int c    = tid & 127;
    int half = tid >> 7;
    size_t base = (size_t)blockIdx.x * 1024;
    float s = 0.f, s2 = 0.f;
    for (int i = 0; i < 512; ++i) {
        float v = conv[(base + (size_t)i * 2 + half) * COUT + c];
        s += v; s2 += v * v;
    }
    __shared__ float r1[256], r2[256];
    r1[tid] = s; r2[tid] = s2;
    __syncthreads();
    if (tid < 128) {
        atomicAdd(&gsum[c], r1[tid] + r1[tid + 128]);
        atomicAdd(&gsq[c],  r2[tid] + r2[tid + 128]);
    }
}

// ---------------------------------------------------------------------------
// K4: fold BN into 1x1 weights. 5 sets: 0=(k,x) 1=(q,y) 2=(k,y) 3=(v,x) 4=(v,y)
//     Wf[set][kc=4][o=128][32] bf16 ; fbias[set][o] = b + sum_c W[o][c]*shift[c]
// ---------------------------------------------------------------------------
__global__ __launch_bounds__(128) void fold_kernel(const float* __restrict__ qw, const float* __restrict__ qb,
                                                   const float* __restrict__ kw, const float* __restrict__ kb,
                                                   const float* __restrict__ vw, const float* __restrict__ vb,
                                                   const float* __restrict__ bnw, const float* __restrict__ bnb,
                                                   const float* __restrict__ sumx, const float* __restrict__ sqx,
                                                   const float* __restrict__ sumy, const float* __restrict__ sqy,
                                                   bf16* __restrict__ Wf, float* __restrict__ fbias) {
    int o = blockIdx.x;
    int s = blockIdx.y;
    int c = threadIdx.x;
    const float *W, *bb, *gs, *gq;
    if (s == 0)      { W = kw; bb = kb; gs = sumx; gq = sqx; }
    else if (s == 1) { W = qw; bb = qb; gs = sumy; gq = sqy; }
    else if (s == 2) { W = kw; bb = kb; gs = sumy; gq = sqy; }
    else if (s == 3) { W = vw; bb = vb; gs = sumx; gq = sqx; }
    else             { W = vw; bb = vb; gs = sumy; gq = sqy; }
    const float invN = 1.f / 65536.f;
    float mean = gs[c] * invN;
    float var  = gq[c] * invN - mean * mean;
    float g    = bnw[c] * rsqrtf(var + EPSV);
    float sh   = bnb[c] - mean * g;
    float w    = W[o * COUT + c];
    Wf[((s * 4 + (c >> 5)) * COUT + o) * 32 + (c & 31)] = (bf16)(w * g);
    __shared__ float red[128];
    red[c] = w * sh;
    __syncthreads();
    for (int off = 64; off; off >>= 1) {
        if (c < off) red[c] += red[c + off];
        __syncthreads();
    }
    if (c == 0) fbias[s * COUT + o] = bb[o] + red[0];
}

// ---------------------------------------------------------------------------
// K5: sum over T: Xsum[b][p] = sum_t conv[(b*4+t)][p]   (p over 131072)
// ---------------------------------------------------------------------------
__global__ void tsum_kernel(const float* __restrict__ conv, float* __restrict__ sum) {
    size_t i = (size_t)blockIdx.x * 256 + threadIdx.x;   // < 16*131072
    size_t b = i >> 17;
    size_t p = i & 131071;
    const float* src = conv + b * 524288 + p;
    sum[i] = src[0] + src[131072] + src[262144] + src[393216];
}

// ---------------------------------------------------------------------------
// K6: 1x1 projection GEMM: out[row][o] = bf16( A[row][:]*Wf + bias*scale )
//     tile 64 rows x 128 cols, K=128 (4 chunks); weights async->LDS
// ---------------------------------------------------------------------------
__global__ __launch_bounds__(256) void proj_kernel(const float* __restrict__ A,
                                                   const bf16* __restrict__ Wf,
                                                   const float* __restrict__ fb,
                                                   bf16* __restrict__ outp,
                                                   float biasScale) {
    int rt = blockIdx.x;
    __shared__ __align__(16) bf16 ldsA[64 * 32];
    __shared__ __align__(16) bf16 ldsB[128 * 32];
    int tid = threadIdx.x, wave = tid >> 5, lane = tid & 31;
    int msub = wave & 3, nbase = (wave >> 2) * 64;
    v8f acc[4];
    for (int j = 0; j < 4; ++j) acc[j] = vzero8();
    const float* ain = A + (size_t)rt * 64 * COUT;
#pragma unroll
    for (int kc = 0; kc < 4; ++kc) {
        const char* wsrc = (const char*)(Wf + kc * 4096);
        async_ld_b128((char*)ldsB + tid * 16,         wsrc + tid * 16);
        async_ld_b128((char*)ldsB + (tid + 256) * 16, wsrc + (tid + 256) * 16);
#pragma unroll
        for (int u = 0; u < 8; ++u) {
            int i = tid + u * 256;
            int m = i >> 5, kk = i & 31;
            ldsA[m * 32 + kk] = (bf16)ain[m * COUT + kc * 32 + kk];
        }
        async_wait0();
        __syncthreads();
        v16bf a = load_frag_a(ldsA, msub * 16);
#pragma unroll
        for (int j = 0; j < 4; ++j) {
            v16bf b = load_frag_b(ldsB, nbase + j * 16);
            WMMA_BF16(acc[j], a, b);
        }
        __syncthreads();
    }
    int mrow = msub * 16 + 8 * (lane >> 4);
    int ncol = lane & 15;
    bf16* op = outp + (size_t)rt * 64 * COUT;
#pragma unroll
    for (int j = 0; j < 4; ++j) {
        int col = nbase + j * 16 + ncol;
        float b = fb[col] * biasScale;
        for (int r = 0; r < 8; ++r)
            op[(mrow + r) * COUT + col] = (bf16)(acc[j][r] + b);
    }
}

// ---------------------------------------------------------------------------
// K7: Gram GEMM per batch: sim[b][c][d] = sum_n KX[b][n][c] * QY[b][n][d], K=1024
//     (staging is a transpose -> element path, cannot use async 16B copies)
// ---------------------------------------------------------------------------
__global__ __launch_bounds__(256) void sim_kernel(const bf16* __restrict__ KX,
                                                  const bf16* __restrict__ QY,
                                                  float* __restrict__ sim) {
    int b = blockIdx.x;
    __shared__ __align__(16) bf16 ldsA[128 * 32];
    __shared__ __align__(16) bf16 ldsB[128 * 32];
    int tid = threadIdx.x, wave = tid >> 5, lane = tid & 31;
    int msub = wave;                 // 0..7
    v8f acc[8];
    for (int j = 0; j < 8; ++j) acc[j] = vzero8();
    const bf16* kxp = KX + (size_t)b * NSP * COUT;
    const bf16* qyp = QY + (size_t)b * NSP * COUT;
    for (int nc = 0; nc < 32; ++nc) {
        int n0 = nc * 32;
#pragma unroll
        for (int u = 0; u < 16; ++u) {
            int i  = tid + u * 256;
            int kk = i >> 7;         // n within chunk
            int c  = i & 127;
            ldsA[c * 32 + kk] = kxp[(n0 + kk) * COUT + c];
            ldsB[c * 32 + kk] = qyp[(n0 + kk) * COUT + c];
        }
        __syncthreads();
        v16bf a = load_frag_a(ldsA, msub * 16);
#pragma unroll
        for (int j = 0; j < 8; ++j) {
            v16bf bb = load_frag_b(ldsB, j * 16);
            WMMA_BF16(acc[j], a, bb);
        }
        __syncthreads();
    }
    int mrow = msub * 16 + 8 * (lane >> 4);
    int ncol = lane & 15;
    float* sp = sim + (size_t)b * COUT * COUT;
#pragma unroll
    for (int j = 0; j < 8; ++j)
        for (int r = 0; r < 8; ++r)
            sp[(mrow + r) * COUT + j * 16 + ncol] = acc[j][r];
}

// ---------------------------------------------------------------------------
// K8: row softmax over 128 values; store transposed as bf16: attT[b][i][j]=att[b][j][i]
// ---------------------------------------------------------------------------
__global__ __launch_bounds__(256) void softmax_kernel(const float* __restrict__ sim,
                                                      bf16* __restrict__ attT) {
    int row  = blockIdx.x * 8 + (threadIdx.x >> 5);   // 0..2047
    int lane = threadIdx.x & 31;
    int b = row >> 7, j = row & 127;
    const float* sp = sim + (size_t)row * COUT;
    float v[4];
    float mx = -3.0e38f;
    for (int k = 0; k < 4; ++k) { v[k] = sp[lane + k * 32]; mx = fmaxf(mx, v[k]); }
    for (int off = 16; off; off >>= 1) mx = fmaxf(mx, __shfl_xor(mx, off, 32));
    float sum = 0.f;
    for (int k = 0; k < 4; ++k) { v[k] = __expf(v[k] - mx); sum += v[k]; }
    for (int off = 16; off; off >>= 1) sum += __shfl_xor(sum, off, 32);
    float inv = 1.f / sum;
    bf16* ap = attT + (size_t)b * COUT * COUT;
    for (int k = 0; k < 4; ++k)
        ap[(lane + k * 32) * COUT + j] = (bf16)(v[k] * inv);
}

// ---------------------------------------------------------------------------
// K9: attention output GEMM per (bt): out[i][n] = sum_j attT[b][i][j]*V[bt][n][j]
//     tile 128 x 128 per block, K=128; both tiles staged via async global->LDS
// ---------------------------------------------------------------------------
__global__ __launch_bounds__(256) void attnout_kernel(const bf16* __restrict__ attT,
                                                      const bf16* __restrict__ V,
                                                      float* __restrict__ outp) {
    int bt = blockIdx.x;     // 0..63
    int nt = blockIdx.y;     // 0..7
    int b  = bt >> 2;
    __shared__ __align__(16) bf16 ldsA[128 * 32];
    __shared__ __align__(16) bf16 ldsB[128 * 32];
    int tid = threadIdx.x, wave = tid >> 5, lane = tid & 31;
    int msub = wave;
    v8f acc[8];
    for (int j = 0; j < 8; ++j) acc[j] = vzero8();
    const bf16* ap = attT + (size_t)b * COUT * COUT;
    const bf16* vp = V + ((size_t)bt * NSP + nt * 128) * COUT;
#pragma unroll
    for (int kc = 0; kc < 4; ++kc) {
        // each tile: 128 rows x 64B; 512 x b128 per tile, 2 per lane
#pragma unroll
        for (int u = 0; u < 2; ++u) {
            int idx = tid + u * 256;          // 0..511
            int r   = idx >> 2;               // row
            int sub = (idx & 3) * 16;         // 16B chunk within 64B row
            async_ld_b128((char*)ldsA + r * 64 + sub,
                          (const char*)(ap + r * COUT + kc * 32) + sub);
            async_ld_b128((char*)ldsB + r * 64 + sub,
                          (const char*)(vp + r * COUT + kc * 32) + sub);
        }
        async_wait0();
        __syncthreads();
        v16bf a = load_frag_a(ldsA, msub * 16);
#pragma unroll
        for (int j = 0; j < 8; ++j) {
            v16bf bb = load_frag_b(ldsB, j * 16);
            WMMA_BF16(acc[j], a, bb);
        }
        __syncthreads();
    }
    int mrow = msub * 16 + 8 * (lane >> 4);
    int ncol = lane & 15;
    float* op = outp + (size_t)bt * COUT * NSP;
#pragma unroll
    for (int j = 0; j < 8; ++j)
        for (int r = 0; r < 8; ++r)
            op[(mrow + r) * NSP + nt * 128 + j * 16 + ncol] = acc[j][r];
}

// ---------------------------------------------------------------------------
// K10: LIF over T (tau=0.5, thresh=1, hard reset).  xa layout [bt][c][n] matches
//      d_out layout [b][t][c][n] elementwise.
// ---------------------------------------------------------------------------
__global__ void lif_kernel(const float* __restrict__ xa, float* __restrict__ outp) {
    size_t i = (size_t)blockIdx.x * 256 + threadIdx.x;   // < 16*131072
    size_t b = i >> 17;
    size_t p = i & 131071;
    float mem = 0.f;
    for (int t = 0; t < 4; ++t) {
        size_t idx = (b * 4 + t) * 131072 + p;
        mem = mem * 0.5f + xa[idx];
        float s = (mem - 1.0f) > 0.f ? 1.f : 0.f;
        outp[idx] = s;
        mem = (1.f - s) * mem;
    }
}

// ---------------------------------------------------------------------------
// Launch
// ---------------------------------------------------------------------------
extern "C" void kernel_launch(void* const* d_in, const int* in_sizes, int n_in,
                              void* d_out, int out_size, void* d_ws, size_t ws_size,
                              hipStream_t stream) {
    const float* x      = (const float*)d_in[0];
    const float* y      = (const float*)d_in[1];
    const float* conv_w = (const float*)d_in[2];
    // conv_b (d_in[3]) is absorbed by BatchNorm -> unused
    const float* bn_w   = (const float*)d_in[4];
    const float* bn_b   = (const float*)d_in[5];
    const float* q_w    = (const float*)d_in[6];
    const float* q_b    = (const float*)d_in[7];
    const float* k_w    = (const float*)d_in[8];
    const float* k_b    = (const float*)d_in[9];
    const float* v_w    = (const float*)d_in[10];
    const float* v_b    = (const float*)d_in[11];
    float* outp = (float*)d_out;

    char* ws = (char*)d_ws;
    size_t off = 0;
    auto alloc = [&](size_t bytes) { char* p = ws + off; off += (bytes + 255) & ~(size_t)255; return p; };
    float* convX = (float*)alloc(33554432);   // [64][1024][128] f32 (reused as xa)
    float* convY = (float*)alloc(33554432);   // reused as ya
    float* Xsum  = (float*)alloc(8388608);    // [16][1024][128]
    float* Ysum  = (float*)alloc(8388608);
    bf16*  KXs   = (bf16*)alloc(4194304);     // [16][1024][128] bf16
    bf16*  QYs   = (bf16*)alloc(4194304);
    bf16*  KYs   = (bf16*)alloc(4194304);
    bf16*  VX    = (bf16*)alloc(16777216);    // [64][1024][128] bf16
    bf16*  VY    = (bf16*)alloc(16777216);
    float* sim1  = (float*)alloc(1048576);    // [16][128][128]
    float* sim2  = (float*)alloc(1048576);
    bf16*  attT1 = (bf16*)alloc(524288);
    bf16*  attT2 = (bf16*)alloc(524288);
    bf16*  Wcv   = (bf16*)alloc(147456);      // [18][128][32]
    bf16*  Wf    = (bf16*)alloc(163840);      // 5 x [4][128][32]
    float* fbias = (float*)alloc(2560);       // 5 x 128
    float* stats = (float*)alloc(2048);       // sumx, sqx, sumy, sqy
    float* sumx = stats, *sqx = stats + 128, *sumy = stats + 256, *sqy = stats + 384;

    hipMemsetAsync(stats, 0, 2048, stream);

    // weight repack
    convw_kernel<<<288, 256, 0, stream>>>(conv_w, Wcv);

    // 3x3 conv (implicit GEMM, WMMA bf16)
    conv3x3_kernel<<<dim3(64, 16), 256, 0, stream>>>(x, Wcv, convX);
    conv3x3_kernel<<<dim3(64, 16), 256, 0, stream>>>(y, Wcv, convY);

    // BN statistics
    stats_kernel<<<64, 256, 0, stream>>>(convX, sumx, sqx);
    stats_kernel<<<64, 256, 0, stream>>>(convY, sumy, sqy);

    // fold BN into projection weights
    fold_kernel<<<dim3(128, 5), 128, 0, stream>>>(q_w, q_b, k_w, k_b, v_w, v_b,
                                                  bn_w, bn_b, sumx, sqx, sumy, sqy,
                                                  Wf, fbias);

    // sum over T
    tsum_kernel<<<8192, 256, 0, stream>>>(convX, Xsum);
    tsum_kernel<<<8192, 256, 0, stream>>>(convY, Ysum);

    // 1x1 projections (WMMA):  sets 0=(k,x) 1=(q,y) 2=(k,y) 3=(v,x) 4=(v,y)
    proj_kernel<<<256, 256, 0, stream>>>(Xsum, Wf + 0 * 16384, fbias + 0 * 128, KXs, 4.f);
    proj_kernel<<<256, 256, 0, stream>>>(Ysum, Wf + 1 * 16384, fbias + 1 * 128, QYs, 4.f);
    proj_kernel<<<256, 256, 0, stream>>>(Ysum, Wf + 2 * 16384, fbias + 2 * 128, KYs, 4.f);
    proj_kernel<<<1024, 256, 0, stream>>>(convX, Wf + 3 * 16384, fbias + 3 * 128, VX, 1.f);
    proj_kernel<<<1024, 256, 0, stream>>>(convY, Wf + 4 * 16384, fbias + 4 * 128, VY, 1.f);

    // Gram matrices (WMMA, K=1024 per batch)
    sim_kernel<<<16, 256, 0, stream>>>(KXs, QYs, sim1);
    sim_kernel<<<16, 256, 0, stream>>>(KYs, QYs, sim2);

    // softmax (+ transpose to attT bf16)
    softmax_kernel<<<256, 256, 0, stream>>>(sim1, attT1);
    softmax_kernel<<<256, 256, 0, stream>>>(sim2, attT2);

    // attention output GEMM (WMMA) — reuse conv buffers as xa / ya
    attnout_kernel<<<dim3(64, 8), 256, 0, stream>>>(attT1, VX, convX);
    attnout_kernel<<<dim3(64, 8), 256, 0, stream>>>(attT2, VY, convY);

    // LIF scan over T
    lif_kernel<<<8192, 256, 0, stream>>>(convX, outp);
    lif_kernel<<<8192, 256, 0, stream>>>(convY, outp + 8388608);
}
